// Transformer_82291573391621
// MI455X (gfx1250) — matrix operand order repaired and checked by reading, not compiled, not run on hardware
//
#include <hip/hip_runtime.h>
#include <hip/hip_bf16.h>
#include <math.h>

#define USE_ASYNC_COPY 1

namespace {

constexpr int BB = 2, SS = 2048, DD = 1024, HH = 16, FF = 4096, LL = 4;
constexpr int HWD = 64;              // head width
constexpr int MTOT = BB * SS;        // 4096 token rows
constexpr float LNEPS = 1e-6f;
constexpr float NEGV = -1e9f;

typedef __attribute__((ext_vector_type(16))) __bf16 bf16x16;
typedef __attribute__((ext_vector_type(8)))  float  f32x8;

union FragU { bf16x16 v; uint4 q[2]; };
union PackU { uint4 q; __bf16 e[8]; };

__device__ __forceinline__ __bf16 f2bf(float f) {
  union { float f; unsigned u; } a; a.f = f;
  unsigned r = a.u + 0x7FFFu + ((a.u >> 16) & 1u);   // round-to-nearest-even
  union { unsigned short s; __bf16 b; } c; c.s = (unsigned short)(r >> 16);
  return c.b;
}

// ---- gfx1250 async global->LDS copy (16B per lane) --------------------------
__device__ __forceinline__ void cp16(void* lds_dst, const void* gsrc) {
#if USE_ASYNC_COPY
  unsigned off = (unsigned)(size_t)lds_dst;          // low 32 bits = LDS offset
  asm volatile("global_load_async_to_lds_b128 %0, %1, off"
               :: "v"(off), "v"(gsrc) : "memory");
#else
  *(uint4*)lds_dst = *(const uint4*)gsrc;
#endif
}
template <int N>
__device__ __forceinline__ void cp_wait_n() {
#if USE_ASYNC_COPY
  asm volatile("s_wait_asynccnt %0" :: "i"(N) : "memory");
#endif
}

__device__ __forceinline__ bf16x16 load_frag(const __bf16* p0, const __bf16* p1) {
  FragU f;
  f.q[0] = *(const uint4*)p0;
  f.q[1] = *(const uint4*)p1;
  return f.v;
}

__device__ __forceinline__ f32x8 wmma_bf16(bf16x16 a, bf16x16 b, f32x8 c) {
  return __builtin_amdgcn_wmma_f32_16x16x32_bf16(false, a, false, b,
                                                 (short)0, c, false, false);
}

// ---- fp32 -> bf16 transposed weight conversion: W[K][N] -> WT[N][K] --------
__global__ void convw_kernel(const float* __restrict__ W, __bf16* __restrict__ WT,
                             int K, int N) {
  size_t i = (size_t)blockIdx.x * 256 + threadIdx.x;
  if (i >= (size_t)K * N) return;
  int k = (int)(i / N), n = (int)(i % N);
  WT[(size_t)n * K + k] = f2bf(W[i]);
}

__global__ void copy_f32_kernel(const float* __restrict__ in, float* __restrict__ out,
                                int n) {
  int i = blockIdx.x * 256 + threadIdx.x;
  if (i < n) out[i] = in[i];
}

// ---- LayerNorm: writes bf16 normalized rows, optionally bf16 copy of x -----
__global__ __launch_bounds__(256)
void ln_kernel(const float* __restrict__ x, const float* __restrict__ g,
               const float* __restrict__ be, __bf16* __restrict__ h,
               __bf16* __restrict__ xb) {
  const int row = blockIdx.x;
  const int tid = threadIdx.x;
  float4 v = ((const float4*)(x + (size_t)row * DD))[tid];
  float s1 = v.x + v.y + v.z + v.w;
  float s2 = v.x * v.x + v.y * v.y + v.z * v.z + v.w * v.w;
  for (int m = 16; m >= 1; m >>= 1) {
    s1 += __shfl_xor(s1, m, 32);
    s2 += __shfl_xor(s2, m, 32);
  }
  __shared__ float a1[8], a2[8];
  if ((tid & 31) == 0) { a1[tid >> 5] = s1; a2[tid >> 5] = s2; }
  __syncthreads();
  s1 = 0.f; s2 = 0.f;
  #pragma unroll
  for (int w = 0; w < 8; ++w) { s1 += a1[w]; s2 += a2[w]; }
  const float mean = s1 * (1.0f / DD);
  const float var  = s2 * (1.0f / DD) - mean * mean;
  const float rs = rsqrtf(var + LNEPS);
  float4 gv = ((const float4*)g)[tid];
  float4 bv = ((const float4*)be)[tid];
  __bf16* hr = h + (size_t)row * DD + tid * 4;
  hr[0] = f2bf((v.x - mean) * rs * gv.x + bv.x);
  hr[1] = f2bf((v.y - mean) * rs * gv.y + bv.y);
  hr[2] = f2bf((v.z - mean) * rs * gv.z + bv.z);
  hr[3] = f2bf((v.w - mean) * rs * gv.w + bv.w);
  if (xb) {
    __bf16* xr = xb + (size_t)row * DD + tid * 4;
    xr[0] = f2bf(v.x); xr[1] = f2bf(v.y); xr[2] = f2bf(v.z); xr[3] = f2bf(v.w);
  }
}

// ---- tiled bf16 WMMA GEMM: C[M][N] = A[M][K] @ WT[N][K]^T + bias -----------
// Double-buffered async->LDS software pipeline (ASYNCcnt counted waits).
// mode 0: bf16 out    mode 1: bf16 gelu(out)
// mode 2: bf16 out, per-head transposed to vt[b*H+h][hw][S]
// mode 3: f32  out = acc + bias + R1 + R2   (residual carry)
__global__ __launch_bounds__(256)
void gemm_kernel(const __bf16* __restrict__ A, const __bf16* __restrict__ WT,
                 const float* __restrict__ bias, void* __restrict__ Cout,
                 const float* __restrict__ R1, const float* __restrict__ R2,
                 int N, int K, int mode) {
  __shared__ __align__(16) __bf16 As[2][128 * 32];
  __shared__ __align__(16) __bf16 Bs[2][128 * 32];
  const int mb = blockIdx.y * 128, nb = blockIdx.x * 128;
  const int tid = threadIdx.x, lane = tid & 31, wid = tid >> 5;
  const int l15 = lane & 15, kh = lane >> 4;
  const int wm = wid & 3, wn = wid >> 2;       // 4x2 wave grid, 32x64 per wave
  f32x8 acc[2][4];
  #pragma unroll
  for (int i = 0; i < 2; ++i)
    #pragma unroll
    for (int j = 0; j < 4; ++j) acc[i][j] = 0.0f;

  // per-thread copy slots: 2 x (16B of A + 16B of B) per tile => 4 async ops
  const int row0 = tid >> 2, ch0 = (tid & 3) << 3;
  const int row1 = (tid + 256) >> 2, ch1 = ((tid + 256) & 3) << 3;

  auto issue_tile = [&](int kt, int buf) {
    const int k0 = kt << 5;
    cp16(&As[buf][row0 * 32 + ch0], A  + (size_t)(mb + row0) * K + k0 + ch0);
    cp16(&Bs[buf][row0 * 32 + ch0], WT + (size_t)(nb + row0) * K + k0 + ch0);
    cp16(&As[buf][row1 * 32 + ch1], A  + (size_t)(mb + row1) * K + k0 + ch1);
    cp16(&Bs[buf][row1 * 32 + ch1], WT + (size_t)(nb + row1) * K + k0 + ch1);
  };

  const int nk = K >> 5;
  issue_tile(0, 0);                              // prologue
  for (int kt = 0; kt < nk; ++kt) {
    const int buf = kt & 1;
    if (kt + 1 < nk) {
      issue_tile(kt + 1, buf ^ 1);               // keep next tile in flight
      cp_wait_n<4>();                            // drain current tile only
    } else {
      cp_wait_n<0>();
    }
    if (kt + 2 < nk)                             // global_prefetch_b8 2 ahead
      __builtin_prefetch(A + (size_t)(mb + row0) * K + ((kt + 2) << 5), 0, 0);
    __syncthreads();

    bf16x16 af[2], bfv[4];
    #pragma unroll
    for (int mi = 0; mi < 2; ++mi) {
      int r = (wm * 32 + mi * 16 + l15) * 32;
      af[mi] = load_frag(&As[buf][r + 8 * kh], &As[buf][r + 16 + 8 * kh]);
    }
    #pragma unroll
    for (int t = 0; t < 4; ++t) {
      int r = (wn * 64 + t * 16 + l15) * 32;
      bfv[t] = load_frag(&Bs[buf][r + 8 * kh], &Bs[buf][r + 16 + 8 * kh]);
    }
    #pragma unroll
    for (int mi = 0; mi < 2; ++mi)
      #pragma unroll
      for (int t = 0; t < 4; ++t)
        acc[mi][t] = wmma_bf16(af[mi], bfv[t], acc[mi][t]);
    __syncthreads();                             // buf may be overwritten next
  }

  #pragma unroll
  for (int mi = 0; mi < 2; ++mi)
    #pragma unroll
    for (int t = 0; t < 4; ++t) {
      const int n  = nb + wn * 64 + t * 16 + l15;
      const int m0 = mb + wm * 32 + mi * 16 + 8 * kh;
      const float bia = bias[n];
      if (mode == 0) {
        __bf16* C = (__bf16*)Cout;
        #pragma unroll
        for (int v = 0; v < 8; ++v)
          C[(size_t)(m0 + v) * N + n] = f2bf(acc[mi][t][v] + bia);
      } else if (mode == 1) {
        __bf16* C = (__bf16*)Cout;
        #pragma unroll
        for (int v = 0; v < 8; ++v) {
          float z = acc[mi][t][v] + bia;
          C[(size_t)(m0 + v) * N + n] =
              f2bf(0.5f * z * (1.0f + erff(z * 0.70710678118f)));
        }
      } else if (mode == 2) {
        __bf16* C = (__bf16*)Cout;
        int h = n >> 6, hw = n & 63;
        int b = m0 >> 11, s0 = m0 & (SS - 1);
        PackU pk;
        #pragma unroll
        for (int v = 0; v < 8; ++v) pk.e[v] = f2bf(acc[mi][t][v] + bia);
        *(uint4*)(C + ((size_t)(b * HH + h) * HWD + hw) * SS + s0) = pk.q;
      } else {
        float* C = (float*)Cout;
        #pragma unroll
        for (int v = 0; v < 8; ++v) {
          size_t ix = (size_t)(m0 + v) * N + n;
          C[ix] = acc[mi][t][v] + bia + R1[ix] + R2[ix];
        }
      }
    }
}

// ---- causal flash attention, bf16 WMMA, online softmax ---------------------
// grid (S/64, B*H), 128 threads; wave w owns 16 query rows.
// K/V tiles double-buffered through async->LDS with counted waits.
__global__ __launch_bounds__(128)
void attn_kernel(const __bf16* __restrict__ qb_, const __bf16* __restrict__ kb_,
                 const __bf16* __restrict__ vt_, const float* __restrict__ x,
                 float* __restrict__ x2) {
  __shared__ __align__(16) __bf16 Kt[2][64 * 64];    // [key][hw]
  __shared__ __align__(16) __bf16 Vt[2][64 * 64];    // [hw][key]
  __shared__ __align__(16) __bf16 Pl[4][16 * 64];    // per-wave [q][key]
  const int qb = blockIdx.x * 64;
  const int bh = blockIdx.y, b = bh >> 4, h = bh & 15;
  const int tid = threadIdx.x, lane = tid & 31, wid = tid >> 5;
  const int l15 = lane & 15, kh = lane >> 4;
  const int qw = qb + wid * 16;

  bf16x16 qf[2];                                     // Q A-frags, hw chunks
  const __bf16* qrow = qb_ + ((size_t)(b * SS + qw + l15)) * DD + h * HWD;
  #pragma unroll
  for (int c = 0; c < 2; ++c)
    qf[c] = load_frag(qrow + 32 * c + 8 * kh, qrow + 32 * c + 16 + 8 * kh);

  f32x8 o[4];
  #pragma unroll
  for (int t = 0; t < 4; ++t) o[t] = 0.0f;
  float mr[8], lr[8];
  #pragma unroll
  for (int v = 0; v < 8; ++v) { mr[v] = -3.0e38f; lr[v] = 0.0f; }

  auto issue_kv = [&](int kblk, int buf) {           // 8 async ops / thread
    const int k0 = kblk << 6;
    #pragma unroll
    for (int i = 0; i < 4; ++i) {
      int u = tid + (i << 7);
      int row = u >> 3, ch = (u & 7) << 3;
      cp16(&Kt[buf][row * 64 + ch],
           kb_ + ((size_t)(b * SS + k0 + row)) * DD + h * HWD + ch);
      cp16(&Vt[buf][row * 64 + ch],
           vt_ + ((size_t)(bh * HWD + row)) * SS + k0 + ch);
    }
  };

  const int nkb = (qb >> 6) + 1;
  issue_kv(0, 0);                                    // prologue
  for (int kblk = 0; kblk < nkb; ++kblk) {
    const int buf = kblk & 1;
    if (kblk + 1 < nkb) {
      issue_kv(kblk + 1, buf ^ 1);
      cp_wait_n<8>();                                // current tile drained
    } else {
      cp_wait_n<0>();
    }
    __syncthreads();
    const int k0 = kblk << 6;

    if (k0 <= qw + 15) {                             // wave has unmasked work
      f32x8 sf[4];
      #pragma unroll
      for (int f = 0; f < 4; ++f) {                  // S = Q @ K^T (16x64)
        f32x8 a = 0.0f;
        #pragma unroll
        for (int c = 0; c < 2; ++c) {
          int r = (f * 16 + l15) * 64 + 32 * c;
          bf16x16 bb = load_frag(&Kt[buf][r + 8 * kh], &Kt[buf][r + 16 + 8 * kh]);
          a = wmma_bf16(qf[c], bb, a);
        }
        sf[f] = a * 0.125f;                          // 1/sqrt(64)
        const int key = k0 + f * 16 + l15;
        #pragma unroll
        for (int v = 0; v < 8; ++v)
          if (key > qw + 8 * kh + v) sf[f][v] += NEGV;
      }
      float mx[8];
      #pragma unroll
      for (int v = 0; v < 8; ++v)
        mx[v] = fmaxf(fmaxf(sf[0][v], sf[1][v]), fmaxf(sf[2][v], sf[3][v]));
      for (int dd = 1; dd < 16; dd <<= 1)
        #pragma unroll
        for (int v = 0; v < 8; ++v) mx[v] = fmaxf(mx[v], __shfl_xor(mx[v], dd, 32));
      float al[8], rsum[8];
      #pragma unroll
      for (int v = 0; v < 8; ++v) {
        float mn = fmaxf(mr[v], mx[v]);
        al[v] = __expf(mr[v] - mn);
        mr[v] = mn;
        rsum[v] = 0.0f;
      }
      #pragma unroll
      for (int f = 0; f < 4; ++f)
        #pragma unroll
        for (int v = 0; v < 8; ++v) {
          float p = __expf(sf[f][v] - mr[v]);
          sf[f][v] = p;
          rsum[v] += p;
        }
      for (int dd = 1; dd < 16; dd <<= 1)
        #pragma unroll
        for (int v = 0; v < 8; ++v) rsum[v] += __shfl_xor(rsum[v], dd, 32);
      #pragma unroll
      for (int v = 0; v < 8; ++v) lr[v] = lr[v] * al[v] + rsum[v];
      #pragma unroll
      for (int t = 0; t < 4; ++t)
        #pragma unroll
        for (int v = 0; v < 8; ++v) o[t][v] *= al[v];

      // reshape P: C-layout -> LDS [q][key] (A-layout source)
      #pragma unroll
      for (int f = 0; f < 4; ++f)
        #pragma unroll
        for (int v = 0; v < 8; ++v)
          Pl[wid][(8 * kh + v) * 64 + f * 16 + l15] = f2bf(sf[f][v]);
      asm volatile("s_wait_dscnt 0" ::: "memory");

      #pragma unroll
      for (int c = 0; c < 2; ++c) {                  // O += P @ V
        int r = l15 * 64 + 32 * c;
        bf16x16 ap = load_frag(&Pl[wid][r + 8 * kh], &Pl[wid][r + 16 + 8 * kh]);
        #pragma unroll
        for (int t = 0; t < 4; ++t) {
          int rv = (t * 16 + l15) * 64 + 32 * c;
          bf16x16 bv = load_frag(&Vt[buf][rv + 8 * kh], &Vt[buf][rv + 16 + 8 * kh]);
          o[t] = wmma_bf16(ap, bv, o[t]);
        }
      }
    }
    __syncthreads();                                 // buf reused next+1 iter
  }

  float inv[8];
  #pragma unroll
  for (int v = 0; v < 8; ++v) inv[v] = 1.0f / lr[v];
  #pragma unroll
  for (int t = 0; t < 4; ++t) {
    const int dcol = h * HWD + t * 16 + l15;
    #pragma unroll
    for (int v = 0; v < 8; ++v) {
      const int m = qw + 8 * kh + v;
      size_t ix = (size_t)(b * SS + m) * DD + dcol;
      x2[ix] = x[ix] + o[t][v] * inv[v];             // x2 = x + attn
    }
  }
}

} // anonymous namespace

extern "C" void kernel_launch(void* const* d_in, const int* in_sizes, int n_in,
                              void* d_out, int out_size, void* d_ws, size_t ws_size,
                              hipStream_t stream) {
  (void)in_sizes; (void)n_in; (void)out_size; (void)ws_size;
  const float* embd  = (const float*)d_in[0];
  const float* wq    = (const float*)d_in[1];
  const float* bq    = (const float*)d_in[2];
  const float* wk    = (const float*)d_in[3];
  const float* bk    = (const float*)d_in[4];
  const float* wv    = (const float*)d_in[5];
  const float* bvp   = (const float*)d_in[6];
  const float* ln1g  = (const float*)d_in[7];
  const float* ln1b  = (const float*)d_in[8];
  const float* w1    = (const float*)d_in[9];
  const float* b1    = (const float*)d_in[10];
  const float* w2    = (const float*)d_in[11];
  const float* b2    = (const float*)d_in[12];
  const float* ln2g  = (const float*)d_in[13];
  const float* ln2b  = (const float*)d_in[14];

  char* ws = (char*)d_ws;
  size_t off = 0;
  auto carve = [&](size_t bytes) -> void* {
    void* p = ws + off;
    off += (bytes + 255) & ~(size_t)255;
    return p;
  };
  __bf16* wqT  = (__bf16*)carve((size_t)DD * DD * 2);
  __bf16* wkT  = (__bf16*)carve((size_t)DD * DD * 2);
  __bf16* wvT  = (__bf16*)carve((size_t)DD * DD * 2);
  __bf16* w1T  = (__bf16*)carve((size_t)DD * FF * 2);
  __bf16* w2T  = (__bf16*)carve((size_t)FF * DD * 2);
  float*  xcur = (float*)carve((size_t)MTOT * DD * 4);
  float*  x2   = (float*)carve((size_t)MTOT * DD * 4);
  __bf16* xbf  = (__bf16*)carve((size_t)MTOT * DD * 2);
  __bf16* hbf  = (__bf16*)carve((size_t)MTOT * DD * 2);
  __bf16* qbf  = (__bf16*)carve((size_t)MTOT * DD * 2);
  __bf16* kbf  = (__bf16*)carve((size_t)MTOT * DD * 2);
  __bf16* vtbf = (__bf16*)carve((size_t)MTOT * DD * 2);   // [B*H][64][S]
  __bf16* ffn1 = (__bf16*)carve((size_t)MTOT * FF * 2);

  const int nTok = MTOT * DD;
  copy_f32_kernel<<<(nTok + 255) / 256, 256, 0, stream>>>(embd, xcur, nTok);

  const dim3 gD(DD / 128, MTOT / 128);   // N=1024 GEMMs
  const dim3 gF(FF / 128, MTOT / 128);   // N=4096 GEMM
  const dim3 gAttn(SS / 64, BB * HH);

  for (int l = 0; l < LL; ++l) {
    const size_t dd = (size_t)DD * DD, df = (size_t)DD * FF;
    convw_kernel<<<(int)((dd + 255) / 256), 256, 0, stream>>>(wq + l * dd, wqT, DD, DD);
    convw_kernel<<<(int)((dd + 255) / 256), 256, 0, stream>>>(wk + l * dd, wkT, DD, DD);
    convw_kernel<<<(int)((dd + 255) / 256), 256, 0, stream>>>(wv + l * dd, wvT, DD, DD);
    convw_kernel<<<(int)((df + 255) / 256), 256, 0, stream>>>(w1 + l * df, w1T, DD, FF);
    convw_kernel<<<(int)((df + 255) / 256), 256, 0, stream>>>(w2 + l * df, w2T, FF, DD);

    // LN1: h = LN(x); also bf16 copy of raw x (q path uses raw x)
    ln_kernel<<<MTOT, 256, 0, stream>>>(xcur, ln1g + l * DD, ln1b + l * DD, hbf, xbf);

    gemm_kernel<<<gD, 256, 0, stream>>>(xbf, wqT, bq + l * DD, qbf, nullptr, nullptr,
                                        DD, DD, 0);
    gemm_kernel<<<gD, 256, 0, stream>>>(hbf, wkT, bk + l * DD, kbf, nullptr, nullptr,
                                        DD, DD, 0);
    gemm_kernel<<<gD, 256, 0, stream>>>(hbf, wvT, bvp + l * DD, vtbf, nullptr, nullptr,
                                        DD, DD, 2);

    attn_kernel<<<gAttn, 128, 0, stream>>>(qbf, kbf, vtbf, xcur, x2);

    // LN2 on x2 -> h2 (reuse hbf)
    ln_kernel<<<MTOT, 256, 0, stream>>>(x2, ln2g + l * DD, ln2b + l * DD, hbf, nullptr);

    gemm_kernel<<<gF, 256, 0, stream>>>(hbf, w1T, b1 + l * FF, ffn1, nullptr, nullptr,
                                        FF, DD, 1);
    // x_next = x + x2 + (ffn1 @ W2 + b2), written in place into xcur
    gemm_kernel<<<gD, 256, 0, stream>>>(ffn1, w2T, b2 + l * DD, xcur, xcur, x2,
                                        DD, FF, 3);
  }

  copy_f32_kernel<<<(nTok + 255) / 256, 256, 0, stream>>>(xcur, (float*)d_out, nTok);
}
